// HandRefinementNetwork_3659312136826
// MI455X (gfx1250) — matrix operand order repaired and checked by reading, not compiled, not run on hardware
//
#include <hip/hip_runtime.h>

// ---------------------------------------------------------------------------
// Problem constants
// ---------------------------------------------------------------------------
#define BB        16
#define LL        64
#define NP        1024
#define HAND      99
#define JJ        21
#define HID       512
#define NLAYERS   4
#define NHEADS    8
#define DH        64
#define SEQ       128            // 2*L
#define ROWS      (BB*SEQ)       // 2048
#define CATR      2273
#define CATP      2304           // CAT padded to multiple of 32
#define OUTN      99
#define OUTNP     128            // 99 padded to multiple of 32 (wave tile is 32 wide)

typedef __attribute__((ext_vector_type(16))) _Float16 v16h;
typedef __attribute__((ext_vector_type(8)))  _Float16 v8h;
typedef __attribute__((ext_vector_type(8)))  float    v8f;

union HU16 { v16h v; v8h h[2]; };

static __device__ __forceinline__ v8f wmma_f16(v16h a, v16h b, v8f c) {
  return __builtin_amdgcn_wmma_f32_16x16x32_f16(
      /*neg_a=*/false, a, /*neg_b=*/false, b,
      /*c_mod=*/(short)0, c, /*reuse_a=*/false, /*reuse_b=*/false);
}

// ---------------------------------------------------------------------------
// Generic batched WMMA GEMM:  C[z] = A[z] (MxK, row major f16) * Bt[z]^T + bias
// Bt is stored transposed: [Npad][Kpad] f16 (row n holds column n of B).
// flags: 1 = ReLU, 2 = store f16 (else f32)
// Requires: M % 32 == 0, Npad % 32 == 0, Kpad % 32 == 0.
// One 32x32 output tile per wave (2x2 WMMA tiles -> 4 WMMAs per K-step with
// A/B fragment reuse, 1 load-wait amortized over 4 matrix ops).
// 8 waves (256 threads) per block.
// ---------------------------------------------------------------------------
__global__ void k_gemm(const _Float16* __restrict__ A, int lda, long long sAz,
                       const _Float16* __restrict__ Bt, int ldb, long long sBz,
                       void* __restrict__ Cv, int ldc, long long sCz,
                       const float* __restrict__ bias,
                       int M, int Npad, int Nreal, int Kpad, int flags) {
  const int tilesN = Npad >> 5;
  const int tilesM = M >> 5;
  const int wave = threadIdx.x >> 5;
  const int lane = threadIdx.x & 31;
  const int tile = blockIdx.x * 8 + wave;
  if (tile >= tilesM * tilesN) return;
  const int tm = tile / tilesN;
  const int tn = tile - tm * tilesN;
  const long long z = blockIdx.y;

  const _Float16* Ab = A + z * sAz;
  const _Float16* Bb = Bt + z * sBz;

  const int hi = lane >> 4;
  const int l15 = lane & 15;
  // A fragment (16x32): lane<16 holds K 0..15,
  // lane>=16 holds K 16..31 -> two contiguous 8-half (16B) chunks.
  const _Float16* arow0 = Ab + (long long)(tm * 32 + l15) * lda + hi * 16;
  const _Float16* arow1 = arow0 + (long long)16 * lda;
  // B fragment (32x16): lane<16 holds K 0..15, lane>=16 holds K 16..31
  // -> one contiguous 16-half (32B) chunk.
  const _Float16* brow0 = Bb + (long long)(tn * 32 + l15) * ldb + hi * 16;
  const _Float16* brow1 = brow0 + (long long)16 * ldb;

  v8f acc00 = {}, acc01 = {}, acc10 = {}, acc11 = {};
  for (int k0 = 0; k0 < Kpad; k0 += 32) {
    if (k0 + 32 < Kpad) {
      __builtin_prefetch(arow0 + k0 + 32, 0, 1);
      __builtin_prefetch(arow1 + k0 + 32, 0, 1);
      __builtin_prefetch(brow0 + k0 + 32, 0, 1);
      __builtin_prefetch(brow1 + k0 + 32, 0, 1);
    }
    HU16 a0, a1;
    a0.h[0] = *(const v8h*)(arow0 + k0);
    a0.h[1] = *(const v8h*)(arow0 + k0 + 8);
    a1.h[0] = *(const v8h*)(arow1 + k0);
    a1.h[1] = *(const v8h*)(arow1 + k0 + 8);
    v16h b0 = *(const v16h*)(brow0 + k0);
    v16h b1 = *(const v16h*)(brow1 + k0);
    acc00 = wmma_f16(a0.v, b0, acc00);
    acc01 = wmma_f16(a0.v, b1, acc01);
    acc10 = wmma_f16(a1.v, b0, acc10);
    acc11 = wmma_f16(a1.v, b1, acc11);
  }

  // C/D 16x16 f32 layout: n = lane&15 ; VGPR r -> m = r + 8*(lane>>4)
  const int cn0 = tn * 32 + l15;
  const int cn1 = cn0 + 16;
  const int m00 = tm * 32 + hi * 8;
  const int m10 = m00 + 16;
  const float bb0 = (bias && cn0 < Nreal) ? bias[cn0] : 0.0f;
  const float bb1 = (bias && cn1 < Nreal) ? bias[cn1] : 0.0f;

  if (flags & 2) {
    _Float16* C = (_Float16*)Cv + z * sCz;
#pragma unroll
    for (int r = 0; r < 8; ++r) {
      float v00 = acc00[r] + bb0, v01 = acc01[r] + bb1;
      float v10 = acc10[r] + bb0, v11 = acc11[r] + bb1;
      if (flags & 1) {
        v00 = fmaxf(v00, 0.0f); v01 = fmaxf(v01, 0.0f);
        v10 = fmaxf(v10, 0.0f); v11 = fmaxf(v11, 0.0f);
      }
      if (cn0 < Nreal) {
        C[(long long)(m00 + r) * ldc + cn0] = (_Float16)v00;
        C[(long long)(m10 + r) * ldc + cn0] = (_Float16)v10;
      }
      if (cn1 < Nreal) {
        C[(long long)(m00 + r) * ldc + cn1] = (_Float16)v01;
        C[(long long)(m10 + r) * ldc + cn1] = (_Float16)v11;
      }
    }
  } else {
    float* C = (float*)Cv + z * sCz;
#pragma unroll
    for (int r = 0; r < 8; ++r) {
      float v00 = acc00[r] + bb0, v01 = acc01[r] + bb1;
      float v10 = acc10[r] + bb0, v11 = acc11[r] + bb1;
      if (flags & 1) {
        v00 = fmaxf(v00, 0.0f); v01 = fmaxf(v01, 0.0f);
        v10 = fmaxf(v10, 0.0f); v11 = fmaxf(v11, 0.0f);
      }
      if (cn0 < Nreal) {
        C[(long long)(m00 + r) * ldc + cn0] = v00;
        C[(long long)(m10 + r) * ldc + cn0] = v10;
      }
      if (cn1 < Nreal) {
        C[(long long)(m00 + r) * ldc + cn1] = v01;
        C[(long long)(m10 + r) * ldc + cn1] = v11;
      }
    }
  }
}

// ---------------------------------------------------------------------------
// Weight pack: W [K][N] f32  ->  Wt [Npad][Kpad] f16 (transposed, zero padded)
// ---------------------------------------------------------------------------
__global__ void k_pack_wT(const float* __restrict__ W, _Float16* __restrict__ Wt,
                          int K, int N, int Kpad, int Npad) {
  int i = blockIdx.x * 256 + threadIdx.x;
  int total = Npad * Kpad;
  if (i >= total) return;
  int n = i / Kpad;
  int k = i - n * Kpad;
  float v = (n < N && k < K) ? W[(long long)k * N + n] : 0.0f;
  Wt[i] = (_Float16)v;
}

// ---------------------------------------------------------------------------
// Preprocess: rotmat, pc transform, contact attention maps, feature concat.
// One block per (b,l); 256 threads = 8 wave32 waves.
// ---------------------------------------------------------------------------
__global__ void k_preprocess(const float* __restrict__ x_lh, const float* __restrict__ x_rh,
                             const float* __restrict__ j_lh, const float* __restrict__ j_rh,
                             const float* __restrict__ mcon, const float* __restrict__ x_obj,
                             const float* __restrict__ pc,
                             _Float16* __restrict__ XL, _Float16* __restrict__ XR) {
  __shared__ float R[9];
  __shared__ float T[3];
  __shared__ float pct[NP * 3];
  const int bl = blockIdx.x;
  const int b = bl >> 6;
  const int t = threadIdx.x;

  if (t == 0) {
    const float* xo = x_obj + (long long)bl * 10;
    float a1x = xo[3], a1y = xo[4], a1z = xo[5];
    float a2x = xo[6], a2y = xo[7], a2z = xo[8];
    float n1 = rsqrtf(a1x * a1x + a1y * a1y + a1z * a1z);
    float b1x = a1x * n1, b1y = a1y * n1, b1z = a1z * n1;
    float dp = b1x * a2x + b1y * a2y + b1z * a2z;
    float c2x = a2x - dp * b1x, c2y = a2y - dp * b1y, c2z = a2z - dp * b1z;
    float n2 = rsqrtf(c2x * c2x + c2y * c2y + c2z * c2z);
    float b2x = c2x * n2, b2y = c2y * n2, b2z = c2z * n2;
    float b3x = b1y * b2z - b1z * b2y;
    float b3y = b1z * b2x - b1x * b2z;
    float b3z = b1x * b2y - b1y * b2x;
    // R = stack([b1,b2,b3], axis=-1): columns are b1,b2,b3
    R[0] = b1x; R[1] = b2x; R[2] = b3x;
    R[3] = b1y; R[4] = b2y; R[5] = b3y;
    R[6] = b1z; R[7] = b2z; R[8] = b3z;
    T[0] = xo[0]; T[1] = xo[1]; T[2] = xo[2];
  }
  __syncthreads();

  _Float16* rowL = XL + (long long)bl * CATP;
  _Float16* rowR = XR + (long long)bl * CATP;

  for (int k = t; k < NP; k += 256) {
    const float* p = pc + ((long long)b * NP + k) * 3;
    float px = p[0], py = p[1], pz = p[2];
    float qx = R[0] * px + R[1] * py + R[2] * pz + T[0];
    float qy = R[3] * px + R[4] * py + R[5] * pz + T[1];
    float qz = R[6] * px + R[7] * py + R[8] * pz + T[2];
    pct[k * 3 + 0] = qx;
    pct[k * 3 + 1] = qy;
    pct[k * 3 + 2] = qz;
    float nm = sqrtf(qx * qx + qy * qy + qz * qz);
    float mv = mcon[(long long)b * NP + k];
    rowL[162 + k] = (_Float16)mv;  rowR[162 + k] = (_Float16)mv;
    rowL[1186 + k] = (_Float16)nm; rowR[1186 + k] = (_Float16)nm;
  }
  for (int c = t; c < HAND; c += 256) {
    rowL[c] = (_Float16)x_lh[(long long)bl * HAND + c];
    rowR[c] = (_Float16)x_rh[(long long)bl * HAND + c];
  }
  for (int c = t; c < 63; c += 256) {
    rowL[99 + c] = (_Float16)j_lh[(long long)bl * 63 + c];
    rowR[99 + c] = (_Float16)j_rh[(long long)bl * 63 + c];
  }
  for (int c = CATR + t; c < CATP; c += 256) {
    rowL[c] = (_Float16)0.0f;
    rowR[c] = (_Float16)0.0f;
  }
  __syncthreads();

  const int wave = t >> 5, lane = t & 31;
  for (int task = wave; task < 2 * JJ; task += 8) {
    const int hand = task / JJ;
    const int j = task - hand * JJ;
    const float* jp = (hand ? j_rh : j_lh) + ((long long)bl * JJ + j) * 3;
    float jx = jp[0], jy = jp[1], jz = jp[2];
    float best = 3.4e38f;
    int bi = 0x7fffffff;
    for (int k = lane; k < NP; k += 32) {
      float dx = jx - pct[k * 3], dy = jy - pct[k * 3 + 1], dz = jz - pct[k * 3 + 2];
      float d2 = dx * dx + dy * dy + dz * dz;
      if (d2 < best || (d2 == best && k < bi)) { best = d2; bi = k; }
    }
    for (int o = 16; o > 0; o >>= 1) {
      float ob = __shfl_down(best, o, 32);
      int oi = __shfl_down(bi, o, 32);
      if (ob < best || (ob == best && oi < bi)) { best = ob; bi = oi; }
    }
    if (lane == 0) {
      float cx = pct[bi * 3], cy = pct[bi * 3 + 1], cz = pct[bi * 3 + 2];
      float dx = jx - cx, dy = jy - cy, dz = jz - cz;
      _Float16* row = hand ? rowR : rowL;
      row[2210 + j * 3 + 0] = (_Float16)expf(-50.0f * dx * dx);
      row[2210 + j * 3 + 1] = (_Float16)expf(-50.0f * dy * dy);
      row[2210 + j * 3 + 2] = (_Float16)expf(-50.0f * dz * dz);
    }
  }
}

// ---------------------------------------------------------------------------
// Elementwise helpers
// ---------------------------------------------------------------------------
__global__ void k_add_pe(float* __restrict__ X) {  // X: (ROWS, HID)
  int i = blockIdx.x * 256 + threadIdx.x;
  if (i >= ROWS * HID) return;
  int c = i & (HID - 1);
  int row = i >> 9;
  int s = row & (SEQ - 1);
  float f = expf(-(float)(2 * (c >> 1)) * (logf(10000.0f) / (float)HID));
  float pf = (float)(s >> 1) * f;   // pe_frame: position s//2 over L
  float pa = (float)(s & 1) * f;    // pe_agent: position s%2 over 2
  float add = (c & 1) ? (cosf(pf) + cosf(pa)) : (sinf(pf) + sinf(pa));
  X[i] += add;
}

__global__ void k_f32_to_f16(const float* __restrict__ in, _Float16* __restrict__ out, int n) {
  int i = blockIdx.x * 256 + threadIdx.x;
  if (i < n) out[i] = (_Float16)in[i];
}

// QKV (ROWS x 1536 f32) -> Qh/Kh [bh][s][d] f16 (Q scaled 1/sqrt(DH)), Vt [bh][d][s]
__global__ void k_split_qkv(const float* __restrict__ QKV, _Float16* __restrict__ Qh,
                            _Float16* __restrict__ Kh, _Float16* __restrict__ Vt) {
  int i = blockIdx.x * 256 + threadIdx.x;  // 2^20 elements
  if (i >= (BB * NHEADS * SEQ * DH)) return;
  int d = i & 63;
  int s = (i >> 6) & 127;
  int h = (i >> 13) & 7;
  int b = i >> 16;
  const float* base = QKV + (long long)(b * SEQ + s) * (3 * HID) + h * DH + d;
  float q = base[0], k = base[HID], v = base[2 * HID];
  int bh = b * NHEADS + h;
  Qh[((long long)bh * SEQ + s) * DH + d] = (_Float16)(q * 0.125f);  // 1/sqrt(64)
  Kh[((long long)bh * SEQ + s) * DH + d] = (_Float16)k;
  Vt[((long long)bh * DH + d) * SEQ + s] = (_Float16)v;
}

__global__ void k_softmax128(const float* __restrict__ S, _Float16* __restrict__ P) {
  __shared__ float red[SEQ];
  int row = blockIdx.x;
  int t = threadIdx.x;
  float v = S[(long long)row * SEQ + t];
  red[t] = v; __syncthreads();
  for (int o = 64; o > 0; o >>= 1) { if (t < o) red[t] = fmaxf(red[t], red[t + o]); __syncthreads(); }
  float m = red[0]; __syncthreads();
  float e = expf(v - m);
  red[t] = e; __syncthreads();
  for (int o = 64; o > 0; o >>= 1) { if (t < o) red[t] += red[t + o]; __syncthreads(); }
  P[(long long)row * SEQ + t] = (_Float16)(e / red[0]);
}

// O [bh][s][d] f32 -> Oh [(b*SEQ+s)][h*DH+d] f16
__global__ void k_merge_heads(const float* __restrict__ O, _Float16* __restrict__ Oh) {
  int i = blockIdx.x * 256 + threadIdx.x;
  if (i >= (BB * NHEADS * SEQ * DH)) return;
  int d = i & 63;
  int s = (i >> 6) & 127;
  int h = (i >> 13) & 7;
  int b = i >> 16;
  Oh[(long long)(b * SEQ + s) * HID + h * DH + d] = (_Float16)O[i];
}

// out = LayerNorm(A + D) * g + beta ; emits f32 and f16 copies. One block/row.
__global__ void k_residual_ln(const float* __restrict__ A, const float* __restrict__ D,
                              const float* __restrict__ g, const float* __restrict__ beta,
                              float* __restrict__ outF, _Float16* __restrict__ outH) {
  __shared__ float red[256];
  int row = blockIdx.x, t = threadIdx.x;
  long long base = (long long)row * HID;
  float v0 = A[base + t] + D[base + t];
  float v1 = A[base + t + 256] + D[base + t + 256];
  red[t] = v0 + v1; __syncthreads();
  for (int o = 128; o > 0; o >>= 1) { if (t < o) red[t] += red[t + o]; __syncthreads(); }
  float mu = red[0] * (1.0f / HID); __syncthreads();
  float d0 = v0 - mu, d1 = v1 - mu;
  red[t] = d0 * d0 + d1 * d1; __syncthreads();
  for (int o = 128; o > 0; o >>= 1) { if (t < o) red[t] += red[t + o]; __syncthreads(); }
  float rs = rsqrtf(red[0] * (1.0f / HID) + 1e-5f);
  float r0 = d0 * rs * g[t] + beta[t];
  float r1 = d1 * rs * g[t + 256] + beta[t + 256];
  outF[base + t] = r0;       outF[base + t + 256] = r1;
  outH[base + t] = (_Float16)r0; outH[base + t + 256] = (_Float16)r1;
}

// ---------------------------------------------------------------------------
// Host orchestration
// ---------------------------------------------------------------------------
static inline void launch_gemm(hipStream_t s,
                               const _Float16* A, int lda, long long sAz,
                               const _Float16* Bt, int ldb, long long sBz,
                               void* C, int ldc, long long sCz,
                               const float* bias, int M, int Npad, int Nreal, int Kpad,
                               int flags, int batches) {
  int tiles = (M / 32) * (Npad / 32);
  dim3 grid((tiles + 7) / 8, batches, 1);
  k_gemm<<<grid, 256, 0, s>>>(A, lda, sAz, Bt, ldb, sBz, C, ldc, sCz,
                              bias, M, Npad, Nreal, Kpad, flags);
}

extern "C" void kernel_launch(void* const* d_in, const int* in_sizes, int n_in,
                              void* d_out, int out_size, void* d_ws, size_t ws_size,
                              hipStream_t stream) {
  (void)in_sizes; (void)n_in; (void)out_size; (void)ws_size;
  const float* x_lhand = (const float*)d_in[0];
  const float* x_rhand = (const float*)d_in[1];
  const float* j_lhand = (const float*)d_in[2];
  const float* j_rhand = (const float*)d_in[3];
  const float* m_contact = (const float*)d_in[4];
  const float* x_obj = (const float*)d_in[5];
  const float* point_cloud = (const float*)d_in[6];
  const float* fc_lw = (const float*)d_in[7];
  const float* fc_lb = (const float*)d_in[8];
  const float* fc_rw = (const float*)d_in[9];
  const float* fc_rb = (const float*)d_in[10];
  const float* out_lw = (const float*)d_in[11];
  const float* out_lb = (const float*)d_in[12];
  const float* out_rw = (const float*)d_in[13];
  const float* out_rb = (const float*)d_in[14];
  const float* Wqkv = (const float*)d_in[15];
  const float* bqkv = (const float*)d_in[16];
  const float* Wo = (const float*)d_in[17];
  const float* bo = (const float*)d_in[18];
  const float* W1 = (const float*)d_in[19];
  const float* b1f = (const float*)d_in[20];
  const float* W2 = (const float*)d_in[21];
  const float* b2f = (const float*)d_in[22];
  const float* ln1_g = (const float*)d_in[23];
  const float* ln1_b = (const float*)d_in[24];
  const float* ln2_g = (const float*)d_in[25];
  const float* ln2_b = (const float*)d_in[26];

  char* wsp = (char*)d_ws;
  size_t off = 0;
  auto alloc = [&](size_t bytes) -> void* {
    size_t a = (off + 255) & ~(size_t)255;
    off = a + bytes;
    return (void*)(wsp + a);
  };

  // f16 weight buffers (transposed [Npad][Kpad])
  _Float16* fcLT  = (_Float16*)alloc((size_t)HID * CATP * 2);
  _Float16* fcRT  = (_Float16*)alloc((size_t)HID * CATP * 2);
  _Float16* WqkvT = (_Float16*)alloc((size_t)NLAYERS * 3 * HID * HID * 2);
  _Float16* WoT   = (_Float16*)alloc((size_t)NLAYERS * HID * HID * 2);
  _Float16* W1T   = (_Float16*)alloc((size_t)NLAYERS * 4 * HID * HID * 2);
  _Float16* W2T   = (_Float16*)alloc((size_t)NLAYERS * HID * 4 * HID * 2);
  _Float16* outLT = (_Float16*)alloc((size_t)OUTNP * HID * 2);
  _Float16* outRT = (_Float16*)alloc((size_t)OUTNP * HID * 2);
  // activations
  _Float16* XL  = (_Float16*)alloc((size_t)(BB * LL) * CATP * 2);
  _Float16* XR  = (_Float16*)alloc((size_t)(BB * LL) * CATP * 2);
  float*    X   = (float*)alloc((size_t)ROWS * HID * 4);
  _Float16* XH  = (_Float16*)alloc((size_t)ROWS * HID * 2);
  float*    QKV = (float*)alloc((size_t)ROWS * 3 * HID * 4);
  _Float16* Qh  = (_Float16*)alloc((size_t)(BB * NHEADS * SEQ * DH) * 2);
  _Float16* Kh  = (_Float16*)alloc((size_t)(BB * NHEADS * SEQ * DH) * 2);
  _Float16* Vt  = (_Float16*)alloc((size_t)(BB * NHEADS * SEQ * DH) * 2);
  float*    SC  = (float*)alloc((size_t)(BB * NHEADS * SEQ) * SEQ * 4);
  _Float16* ATT = (_Float16*)alloc((size_t)(BB * NHEADS * SEQ) * SEQ * 2);
  float*    OB  = (float*)alloc((size_t)(BB * NHEADS * SEQ * DH) * 4);
  _Float16* OH  = (_Float16*)alloc((size_t)ROWS * HID * 2);
  float*    AO  = (float*)alloc((size_t)ROWS * HID * 4);
  float*    HB  = (float*)alloc((size_t)ROWS * HID * 4);
  _Float16* HH  = (_Float16*)alloc((size_t)ROWS * HID * 2);
  _Float16* FFH = (_Float16*)alloc((size_t)ROWS * 4 * HID * 2);
  float*    FO  = (float*)alloc((size_t)ROWS * HID * 4);

  auto pack = [&](const float* W, _Float16* Wt, int K, int N, int Kpad, int Npad) {
    int total = Npad * Kpad;
    k_pack_wT<<<(total + 255) / 256, 256, 0, stream>>>(W, Wt, K, N, Kpad, Npad);
  };

  // --- pack all weights to f16 transposed
  pack(fc_lw, fcLT, CATR, HID, CATP, HID);
  pack(fc_rw, fcRT, CATR, HID, CATP, HID);
  for (int i = 0; i < NLAYERS; ++i) {
    pack(Wqkv + (size_t)i * HID * 3 * HID, WqkvT + (size_t)i * 3 * HID * HID, HID, 3 * HID, HID, 3 * HID);
    pack(Wo   + (size_t)i * HID * HID,     WoT   + (size_t)i * HID * HID,     HID, HID,     HID, HID);
    pack(W1   + (size_t)i * HID * 4 * HID, W1T   + (size_t)i * 4 * HID * HID, HID, 4 * HID, HID, 4 * HID);
    pack(W2   + (size_t)i * 4 * HID * HID, W2T   + (size_t)i * HID * 4 * HID, 4 * HID, HID, 4 * HID, HID);
  }
  pack(out_lw, outLT, HID, OUTN, HID, OUTNP);
  pack(out_rw, outRT, HID, OUTN, HID, OUTNP);

  // --- preprocess: feature rows XL/XR (f16, K-padded)
  k_preprocess<<<BB * LL, 256, 0, stream>>>(x_lhand, x_rhand, j_lhand, j_rhand,
                                            m_contact, x_obj, point_cloud, XL, XR);

  // --- input projections, interleaved into X (row s=2l -> left, s=2l+1 -> right)
  launch_gemm(stream, XL, CATP, 0, fcLT, CATP, 0, X,        2 * HID, 0, fc_lb,
              BB * LL, HID, HID, CATP, 0, 1);
  launch_gemm(stream, XR, CATP, 0, fcRT, CATP, 0, X + HID,  2 * HID, 0, fc_rb,
              BB * LL, HID, HID, CATP, 0, 1);

  // --- positional encodings, then f16 copy of X
  k_add_pe<<<(ROWS * HID + 255) / 256, 256, 0, stream>>>(X);
  k_f32_to_f16<<<(ROWS * HID + 255) / 256, 256, 0, stream>>>(X, XH, ROWS * HID);

  // --- transformer layers
  for (int i = 0; i < NLAYERS; ++i) {
    // QKV projection
    launch_gemm(stream, XH, HID, 0, WqkvT + (size_t)i * 3 * HID * HID, HID, 0,
                QKV, 3 * HID, 0, bqkv + (size_t)i * 3 * HID,
                ROWS, 3 * HID, 3 * HID, HID, 0, 1);
    k_split_qkv<<<(BB * NHEADS * SEQ * DH + 255) / 256, 256, 0, stream>>>(QKV, Qh, Kh, Vt);

    // scores = (Q * 1/sqrt(DH)) @ K^T  (batched over 128 (b,h))
    launch_gemm(stream, Qh, DH, (long long)SEQ * DH,
                Kh, DH, (long long)SEQ * DH,
                SC, SEQ, (long long)SEQ * SEQ,
                nullptr, SEQ, SEQ, SEQ, DH, 0, BB * NHEADS);
    k_softmax128<<<BB * NHEADS * SEQ, SEQ, 0, stream>>>(SC, ATT);

    // O = att @ V  (V stored transposed so it is the Bt operand directly)
    launch_gemm(stream, ATT, SEQ, (long long)SEQ * SEQ,
                Vt, SEQ, (long long)DH * SEQ,
                OB, DH, (long long)SEQ * DH,
                nullptr, SEQ, DH, DH, SEQ, 0, BB * NHEADS);
    k_merge_heads<<<(BB * NHEADS * SEQ * DH + 255) / 256, 256, 0, stream>>>(OB, OH);

    // output projection of attention
    launch_gemm(stream, OH, HID, 0, WoT + (size_t)i * HID * HID, HID, 0,
                AO, HID, 0, bo + (size_t)i * HID, ROWS, HID, HID, HID, 0, 1);

    // h = LN(x + attn) ; emits HB (f32) + HH (f16)
    k_residual_ln<<<ROWS, 256, 0, stream>>>(X, AO, ln1_g + (size_t)i * HID,
                                            ln1_b + (size_t)i * HID, HB, HH);

    // FF: relu(h @ W1 + b1) stored directly as f16, then @ W2 + b2
    launch_gemm(stream, HH, HID, 0, W1T + (size_t)i * 4 * HID * HID, HID, 0,
                FFH, 4 * HID, 0, b1f + (size_t)i * 4 * HID,
                ROWS, 4 * HID, 4 * HID, HID, /*relu|f16*/ 3, 1);
    launch_gemm(stream, FFH, 4 * HID, 0, W2T + (size_t)i * HID * 4 * HID, 4 * HID, 0,
                FO, HID, 0, b2f + (size_t)i * HID, ROWS, HID, HID, 4 * HID, 0, 1);

    // x = LN(h + ff) ; emits X (f32) + XH (f16) for next layer / output heads
    k_residual_ln<<<ROWS, 256, 0, stream>>>(HB, FO, ln2_g + (size_t)i * HID,
                                            ln2_b + (size_t)i * HID, X, XH);
  }

  // --- output heads: even rows -> out_l, odd rows -> out_r (row stride 2*HID)
  float* out_l = (float*)d_out;
  float* out_r = out_l + (size_t)(BB * LL) * OUTN;
  launch_gemm(stream, XH, 2 * HID, 0, outLT, HID, 0, out_l, OUTN, 0, out_lb,
              BB * LL, OUTNP, OUTN, HID, 0, 1);
  launch_gemm(stream, XH + HID, 2 * HID, 0, outRT, HID, 0, out_r, OUTN, 0, out_rb,
              BB * LL, OUTNP, OUTN, HID, 0, 1);
}
